// AVWDCRNN_74354473828679
// MI455X (gfx1250) — matrix operand order repaired and checked by reading, not compiled, not run on hardware
//
#include <hip/hip_runtime.h>
#include <hip/hip_bf16.h>
#include <math.h>

// ---------------------------------------------------------------------------
// Problem dims (compile-time)
// ---------------------------------------------------------------------------
#define B_ 32
#define T_ 12
#define N_ 1024
#define DIN_ 2
#define DOUT_ 64
#define D_ 10
#define SK_ 16
#define TK_ 4
#define KP_ 96          // padded K (66 -> 96, multiple of 32 for WMMA bf16)

typedef __attribute__((ext_vector_type(16))) __bf16 v16bf;
typedef __attribute__((ext_vector_type(8)))  __bf16 v8bf;
typedef __attribute__((ext_vector_type(8)))  float  v8f;
typedef __attribute__((ext_vector_type(8)))  unsigned short v8us;
typedef __attribute__((ext_vector_type(4)))  unsigned int   u32x4;
typedef __attribute__((ext_vector_type(8)))  int            i32x8;
typedef __attribute__((ext_vector_type(4)))  int            i32x4;

#if __has_builtin(__builtin_amdgcn_tensor_load_to_lds)
#define HAVE_TDM 1
#endif

// float -> bf16 bits, round-to-nearest-even
static __device__ __forceinline__ unsigned short f2bf(float f) {
    union { float f; unsigned int u; } v; v.f = f;
    unsigned int u = v.u;
    unsigned int r = (u + 0x7FFFu + ((u >> 16) & 1u)) >> 16;
    return (unsigned short)r;
}

// ---------------------------------------------------------------------------
// Precompute: combined per-node weights  C_n = W_n * cW1^T   (bf16, [N][OC][96])
// and combined bias bb_n = cW1*b_n + cW2*emb_n + cb          (f32,  [N][OC])
// where W_n = sum_d emb[n,d] * wpool[d]  ([66][OC]),  b_n = emb[n] @ bpool.
// ---------------------------------------------------------------------------
template<int OC>
__global__ void prep_weights(const float* __restrict__ emb,
                             const float* __restrict__ wpool,   // [D][66][OC]
                             const float* __restrict__ bpool,   // [D][OC]
                             const float* __restrict__ cW,      // [OC][OC+D]
                             const float* __restrict__ cb,      // [OC]
                             unsigned short* __restrict__ Wout, // [N][OC][KP_]
                             float* __restrict__ bout)          // [N][OC]
{
    const int n = blockIdx.x;
    const int o = threadIdx.x;          // 0..OC-1
    __shared__ float Wlds[66 * OC];
    __shared__ float blds[OC];

    float e[D_];
#pragma unroll
    for (int d = 0; d < D_; ++d) e[d] = emb[n * D_ + d];

    for (int idx = o; idx < 66 * OC; idx += OC) {
        const int i  = idx / OC;
        const int oo = idx % OC;
        float acc = 0.f;
#pragma unroll
        for (int d = 0; d < D_; ++d) acc += e[d] * wpool[(d * 66 + i) * OC + oo];
        Wlds[idx] = acc;
    }
    float bn = 0.f;
#pragma unroll
    for (int d = 0; d < D_; ++d) bn += e[d] * bpool[d * OC + o];
    blds[o] = bn;
    __syncthreads();

    const float* cWrow = cW + (size_t)o * (OC + D_);
    unsigned short* wrow = Wout + ((size_t)n * OC + o) * KP_;
    for (int i = 0; i < 66; ++i) {
        float acc = 0.f;
        for (int op = 0; op < OC; ++op) acc += Wlds[i * OC + op] * cWrow[op];
        wrow[i] = f2bf(acc);
    }
    for (int i = 66; i < KP_; ++i) wrow[i] = 0;   // zero K padding

    float bacc = cb[o];
    for (int op = 0; op < OC; ++op) bacc += cWrow[op] * blds[op];
#pragma unroll
    for (int d = 0; d < D_; ++d) bacc += cWrow[OC + d] * e[d];
    bout[(size_t)n * OC + o] = bacc;
}

// ---------------------------------------------------------------------------
// Init the persistent packed bf16 A matrix: acat[b][n][0:2]=x_0, rest = 0
// (cols 2..65 carry state / z*state, cols 66..95 are permanent zero K-padding)
// ---------------------------------------------------------------------------
__global__ __launch_bounds__(256)
void init_acat(const float* __restrict__ x, unsigned short* __restrict__ acat)
{
    const int idx = blockIdx.x * 256 + threadIdx.x;     // B*N*96
    const int k = idx % KP_;
    const int n = (idx / KP_) & (N_ - 1);
    const int b = idx / (KP_ * N_);
    unsigned short v = 0;
    if (k < DIN_) v = f2bf(x[(((size_t)b * T_ + 0) * N_ + n) * DIN_ + k]);
    acat[idx] = v;
}

// ---------------------------------------------------------------------------
// Per-node WMMA GEMM:  y[b][n][:] = acat[b][n][:96] · C_n + bb_n
// One block per node, 256 threads = 8 wave32.  A = [32 x 96] bf16 staged into
// LDS by the Tensor Data Mover (one 2-D D# per block), B = C_n bf16 stored
// N-major [OC][96], C/D f32.  16x16 tiles, 3 WMMA (K=32) each, uniform loop.
// ---------------------------------------------------------------------------
template<int OUTC>
__global__ __launch_bounds__(256)
void gemm_node(const unsigned short* __restrict__ acat, // [B][N][96] bf16
               const unsigned short* __restrict__ W,    // [N][OUTC][96] bf16
               const float* __restrict__ bias,          // [N][OUTC]
               float* __restrict__ yout)                // [B][N][OUTC]
{
    const int n = blockIdx.x;
    __shared__ __align__(16) unsigned short Alds[B_ * KP_];
    const int tid  = threadIdx.x;
    const int wave = tid >> 5;
    const int lane = tid & 31;
    const int half = lane >> 4;
    const int l15  = lane & 15;
    constexpr int NT = OUTC / 16;

    // Warm the weight stream while the A tile is in flight
    __builtin_prefetch(W + ((size_t)n * OUTC + (wave % NT) * 16 + l15) * KP_, 0, 1);

#ifdef HAVE_TDM
    if (tid == 0) {
        // Tensor DMA: 2-D tile, 32 rows x 48 dwords, row stride N*96*2 bytes.
        const unsigned long long gaddr =
            (unsigned long long)(const void*)(acat + (size_t)n * KP_);
        const unsigned lds_base =
            (unsigned)(unsigned long long)(const void*)&Alds[0]; // aperture low32 = LDS offset
        u32x4 g0;
        g0[0] = 1u;                                   // count=1 (valid user D#)
        g0[1] = lds_base;                             // lds_addr
        g0[2] = (unsigned)gaddr;                      // global_addr[31:0]
        g0[3] = ((unsigned)(gaddr >> 32) & 0x01FFFFFFu) | 0x80000000u; // addr[56:32]|type=2
        i32x8 g1;
        g1[0] = (2 << 16);                 // data_size=4B; no mask/pad/iterate
        g1[1] = (48 << 16);                // tensor_dim0 = 48 dwords (bits 63:48->[31:16])
        g1[2] = 0;                         // tensor_dim0 hi | tensor_dim1 lo
        g1[3] = 0x10 | (48 << 16);         // tensor_dim1 = 1<<20 rows | tile_dim0 = 48
        g1[4] = 32;                        // tile_dim1 = 32 rows, tile_dim2 = 0
        g1[5] = (int)(N_ * KP_ / 2);       // tensor_dim0_stride = 49152 dwords
        g1[6] = 0;
        g1[7] = 0;
        i32x4 gz = {0, 0, 0, 0};
#if __clang_major__ >= 23
        i32x8 gz8 = {0, 0, 0, 0, 0, 0, 0, 0};
        __builtin_amdgcn_tensor_load_to_lds(g0, g1, gz, gz, gz8, 0);
#else
        __builtin_amdgcn_tensor_load_to_lds(g0, g1, gz, gz, 0);
#endif
        __builtin_amdgcn_s_wait_tensorcnt(0);
    }
#else
    // Fallback: cooperative vectorized copy (12 x 16B chunks per row)
    for (int idx = tid; idx < B_ * (KP_ / 8); idx += 256) {
        const int b = idx / (KP_ / 8);
        const int q = idx % (KP_ / 8);
        *(v8us*)&Alds[b * KP_ + q * 8] =
            *(const v8us*)&acat[((size_t)b * N_ + n) * KP_ + q * 8];
    }
#endif
    __syncthreads();

    constexpr int ITERS = (2 * NT) / 8;     // compile-time uniform trip count
#pragma unroll
    for (int it = 0; it < ITERS; ++it) {
        const int tile = wave + it * 8;
        const int mt   = tile / NT;         // batch-row tile
        const int nt   = tile % NT;         // output-channel tile
        const int col  = nt * 16 + l15;

        const float bv = bias[(size_t)n * OUTC + col];
        v8f acc = { bv, bv, bv, bv, bv, bv, bv, bv };

        const unsigned short* wrow = W + ((size_t)n * OUTC + col) * KP_;
        const unsigned short* arow = Alds + (mt * 16 + l15) * KP_;

#pragma unroll
        for (int kc = 0; kc < 3; ++kc) {
            // A 16x32 bf16 fragment: lanes 0-15 K{0..7,16..23}, lanes 16-31 +8
            const int ka = kc * 32 + half * 8;
            v8bf alo = *(const v8bf*)(arow + ka);
            v8bf ahi = *(const v8bf*)(arow + ka + 16);
            v16bf av = __builtin_shufflevector(alo, ahi,
                0,1,2,3,4,5,6,7,8,9,10,11,12,13,14,15);
            // B 32x16 bf16 fragment: lanes 0-15 K{0..7,16..23}, lanes 16-31 +8
            const int kb = kc * 32 + half * 8;
            v8bf blo = *(const v8bf*)(wrow + kb);
            v8bf bhi = *(const v8bf*)(wrow + kb + 16);
            v16bf bvv = __builtin_shufflevector(blo, bhi,
                0,1,2,3,4,5,6,7,8,9,10,11,12,13,14,15);
            acc = __builtin_amdgcn_wmma_f32_16x16x32_bf16(
                false, av, false, bvv, (short)0, acc, false, false);
        }
        // D layout: VGPR i -> row i (lanes 0-15) / row 8+i (lanes 16-31)
#pragma unroll
        for (int i = 0; i < 8; ++i) {
            const int row = mt * 16 + half * 8 + i;   // batch index
            yout[((size_t)row * N_ + n) * OUTC + col] = acc[i];
        }
    }
}

// ---------------------------------------------------------------------------
// Spatial hyperedge reduction: es[nb][s][c] = sum_n Hs[n][s] * y[nb][n][c]
// ---------------------------------------------------------------------------
template<int C>
__global__ void hyper_es(const float* __restrict__ y,   // [NB][N][C]
                         const float* __restrict__ Hs,  // [N][SK]
                         float* __restrict__ es)        // [NB][SK][C]
{
    const int nb = blockIdx.x;
    const int c  = threadIdx.x;
    __shared__ float hs[512 * SK_];

    float acc[SK_];
#pragma unroll
    for (int s = 0; s < SK_; ++s) acc[s] = 0.f;

    for (int chunk = 0; chunk < 2; ++chunk) {
        __syncthreads();
        for (int idx = c; idx < 512 * SK_; idx += C)
            hs[idx] = Hs[chunk * 512 * SK_ + idx];
        __syncthreads();
        for (int n = 0; n < 512; ++n) {
            const float yv = y[((size_t)nb * N_ + chunk * 512 + n) * C + c];
#pragma unroll
            for (int s = 0; s < SK_; ++s) acc[s] += hs[n * SK_ + s] * yv;
        }
    }
#pragma unroll
    for (int s = 0; s < SK_; ++s)
        es[((size_t)nb * SK_ + s) * C + c] = acc[s];
}

// ---------------------------------------------------------------------------
// Gate pointwise: zr = sigmoid(gam*(y*(1+sg) + Hs·es) + bet);
// writes bf16 z*state into acat cols 2..65 (x cols / padding untouched), r.
// ---------------------------------------------------------------------------
__global__ __launch_bounds__(256)
void gate_pointwise(const float* __restrict__ ybuf,  // [B][N][128]
                    const float* __restrict__ es,    // [B][SK][128]
                    const float* __restrict__ Hs,    // [N][SK]
                    const float* __restrict__ ht,    // [1][TK]
                    const float* __restrict__ gam,   // [128]
                    const float* __restrict__ bet,   // [128]
                    const float* __restrict__ state, // [B][N][64]
                    unsigned short* __restrict__ acat, // [B][N][96] bf16
                    float* __restrict__ rbuf)        // [B][N][64]
{
    const int idx = blockIdx.x * 256 + threadIdx.x;   // B*N*64 threads
    const int c = idx & 63;
    const int n = (idx >> 6) & (N_ - 1);
    const int b = idx >> 16;

    float sg = 0.f;
#pragma unroll
    for (int k = 0; k < TK_; ++k) { const float h = ht[k]; sg += h * h; }

    const float y0 = ybuf[((size_t)b * N_ + n) * 128 + c];
    const float y1 = ybuf[((size_t)b * N_ + n) * 128 + 64 + c];
    float sp0 = 0.f, sp1 = 0.f;
#pragma unroll
    for (int s = 0; s < SK_; ++s) {
        const float hv = Hs[n * SK_ + s];
        sp0 += hv * es[((size_t)b * SK_ + s) * 128 + c];
        sp1 += hv * es[((size_t)b * SK_ + s) * 128 + 64 + c];
    }
    const float v0 = gam[c]      * (y0 * (1.f + sg) + sp0) + bet[c];
    const float v1 = gam[64 + c] * (y1 * (1.f + sg) + sp1) + bet[64 + c];
    const float z = 1.f / (1.f + expf(-v0));
    const float r = 1.f / (1.f + expf(-v1));
    const float st = state[((size_t)b * N_ + n) * 64 + c];
    acat[((size_t)b * N_ + n) * KP_ + DIN_ + c] = f2bf(z * st);
    rbuf[((size_t)b * N_ + n) * 64 + c] = r;
}

// ---------------------------------------------------------------------------
// Update pointwise: hc = tanh(...); h = r*state + (1-r)*hc.
// Writes f32 state (in place), seq (=d_out), and preps next step's A matrix:
// bf16 h into acat cols 2..65 and x_{t+1} into cols 0..1.
// ---------------------------------------------------------------------------
__global__ __launch_bounds__(256)
void upd_pointwise(const float* __restrict__ ybuf,  // [B][N][64]
                   const float* __restrict__ es,    // [B][SK][64]
                   const float* __restrict__ Hs,    // [N][SK]
                   const float* __restrict__ ht,    // [1][TK]
                   const float* __restrict__ gam,   // [64]
                   const float* __restrict__ bet,   // [64]
                   const float* __restrict__ rbuf,  // [B][N][64]
                   float* __restrict__ state,       // [B][N][64] in-place
                   float* __restrict__ seq,         // [B][T][N][64] (= d_out)
                   unsigned short* __restrict__ acat, // [B][N][96] bf16
                   const float* __restrict__ x,     // [B][T][N][2]
                   int t)
{
    const int idx = blockIdx.x * 256 + threadIdx.x;
    const int c = idx & 63;
    const int n = (idx >> 6) & (N_ - 1);
    const int b = idx >> 16;

    float su = 0.f;
#pragma unroll
    for (int k = 0; k < TK_; ++k) { const float h = ht[k]; su += h * h; }

    const float y = ybuf[((size_t)b * N_ + n) * 64 + c];
    float sp = 0.f;
#pragma unroll
    for (int s = 0; s < SK_; ++s)
        sp += Hs[n * SK_ + s] * es[((size_t)b * SK_ + s) * 64 + c];

    const float v  = gam[c] * (y * (1.f + su) + sp) + bet[c];
    const float hc = tanhf(v);
    const float r  = rbuf[((size_t)b * N_ + n) * 64 + c];
    const float st = state[((size_t)b * N_ + n) * 64 + c];
    const float h  = r * st + (1.f - r) * hc;
    state[((size_t)b * N_ + n) * 64 + c] = h;
    seq[(((size_t)b * T_ + t) * N_ + n) * 64 + c] = h;

    const size_t base96 = ((size_t)b * N_ + n) * KP_;
    acat[base96 + DIN_ + c] = f2bf(h);
    if (c == 0 && t + 1 < T_) {
        acat[base96 + 0] = f2bf(x[(((size_t)b * T_ + (t + 1)) * N_ + n) * DIN_ + 0]);
        acat[base96 + 1] = f2bf(x[(((size_t)b * T_ + (t + 1)) * N_ + n) * DIN_ + 1]);
    }
}

// ---------------------------------------------------------------------------
// Final combine: out = lgam*(seq + lhs·es_l + (lht·lhtT)·seq) + lbet, in place.
// ---------------------------------------------------------------------------
__global__ __launch_bounds__(256)
void final_combine(float* __restrict__ out,          // [B][T][N][64] in place
                   const float* __restrict__ esl,    // [B*T][SK][64]
                   const float* __restrict__ lhs,    // [N][SK]
                   const float* __restrict__ lht,    // [T][TK]
                   const float* __restrict__ lgam,
                   const float* __restrict__ lbet)
{
    const int idx = blockIdx.x * 256 + threadIdx.x;
    const int c = idx & 63;
    const int n = (idx >> 6) & (N_ - 1);
    const int b = idx >> 16;

    float sv[T_];
#pragma unroll
    for (int t = 0; t < T_; ++t)
        sv[t] = out[(((size_t)b * T_ + t) * N_ + n) * 64 + c];

    float a[TK_] = {0.f, 0.f, 0.f, 0.f};   // a[k] = sum_t lht[t,k] * seq[t]
#pragma unroll
    for (int t = 0; t < T_; ++t)
#pragma unroll
        for (int k = 0; k < TK_; ++k) a[k] += lht[t * TK_ + k] * sv[t];

    float hsr[SK_];
#pragma unroll
    for (int s = 0; s < SK_; ++s) hsr[s] = lhs[n * SK_ + s];

    const float g = lgam[c], be = lbet[c];
#pragma unroll
    for (int t = 0; t < T_; ++t) {
        float sp = 0.f;
#pragma unroll
        for (int s = 0; s < SK_; ++s)
            sp += hsr[s] * esl[(((size_t)b * T_ + t) * SK_ + s) * 64 + c];
        float tp = 0.f;
#pragma unroll
        for (int k = 0; k < TK_; ++k) tp += lht[t * TK_ + k] * a[k];
        out[(((size_t)b * T_ + t) * N_ + n) * 64 + c] = g * (sv[t] + sp + tp) + be;
    }
}

// ---------------------------------------------------------------------------
// Host-side launch
// ---------------------------------------------------------------------------
extern "C" void kernel_launch(void* const* d_in, const int* in_sizes, int n_in,
                              void* d_out, int out_size, void* d_ws, size_t ws_size,
                              hipStream_t stream) {
    const float* x       = (const float*)d_in[0];
    const float* init_st = (const float*)d_in[1];
    const float* emb     = (const float*)d_in[2];
    const float* wpool_g = (const float*)d_in[3];
    const float* bpool_g = (const float*)d_in[4];
    const float* cW_g    = (const float*)d_in[5];
    const float* cb_g    = (const float*)d_in[6];
    const float* hs_g    = (const float*)d_in[7];
    const float* ht_g    = (const float*)d_in[8];
    const float* gam_g   = (const float*)d_in[9];
    const float* bet_g   = (const float*)d_in[10];
    const float* wpool_u = (const float*)d_in[11];
    const float* bpool_u = (const float*)d_in[12];
    const float* cW_u    = (const float*)d_in[13];
    const float* cb_u    = (const float*)d_in[14];
    const float* hs_u    = (const float*)d_in[15];
    const float* ht_u    = (const float*)d_in[16];
    const float* gam_u   = (const float*)d_in[17];
    const float* bet_u   = (const float*)d_in[18];
    const float* lhs     = (const float*)d_in[19];
    const float* lht     = (const float*)d_in[20];
    const float* lgam    = (const float*)d_in[21];
    const float* lbet    = (const float*)d_in[22];

    float* out   = (float*)d_out;                         // [B][T][N][64]
    float* state = out + (size_t)B_ * T_ * N_ * 64;       // "last" slot = live state

    // Workspace carve-up
    char* ws = (char*)d_ws;
    size_t off = 0;
    auto carve = [&](size_t bytes) -> void* {
        void* p = ws + off;
        off += (bytes + 255) & ~(size_t)255;
        return p;
    };
    unsigned short* Wg   = (unsigned short*)carve((size_t)N_ * 128 * KP_ * 2);
    unsigned short* Wu   = (unsigned short*)carve((size_t)N_ * 64  * KP_ * 2);
    float* bbg   = (float*)carve((size_t)N_ * 128 * 4);
    float* bbu   = (float*)carve((size_t)N_ * 64 * 4);
    float* ybuf  = (float*)carve((size_t)B_ * N_ * 128 * 4);
    float* esbuf = (float*)carve((size_t)B_ * T_ * SK_ * 64 * 4);
    float* rbuf  = (float*)carve((size_t)B_ * N_ * 64 * 4);
    unsigned short* acat = (unsigned short*)carve((size_t)B_ * N_ * KP_ * 2);

    // state <- init_state
    hipMemcpyAsync(state, init_st, (size_t)B_ * N_ * 64 * sizeof(float),
                   hipMemcpyDeviceToDevice, stream);

    // Fold cW into per-node weight pools; pack A matrix for step 0
    prep_weights<128><<<N_, 128, 0, stream>>>(emb, wpool_g, bpool_g, cW_g, cb_g, Wg, bbg);
    prep_weights<64> <<<N_, 64,  0, stream>>>(emb, wpool_u, bpool_u, cW_u, cb_u, Wu, bbu);
    init_acat<<<(B_ * N_ * KP_) / 256, 256, 0, stream>>>(x, acat);

    for (int t = 0; t < T_; ++t) {
        // gate branch
        gemm_node<128><<<N_, 256, 0, stream>>>(acat, Wg, bbg, ybuf);
        hyper_es<128> <<<B_, 128, 0, stream>>>(ybuf, hs_g, esbuf);
        gate_pointwise<<<(B_ * N_ * 64) / 256, 256, 0, stream>>>(
            ybuf, esbuf, hs_g, ht_g, gam_g, bet_g, state, acat, rbuf);
        // update branch
        gemm_node<64><<<N_, 256, 0, stream>>>(acat, Wu, bbu, ybuf);
        hyper_es<64> <<<B_, 64, 0, stream>>>(ybuf, hs_u, esbuf);
        upd_pointwise<<<(B_ * N_ * 64) / 256, 256, 0, stream>>>(
            ybuf, esbuf, hs_u, ht_u, gam_u, bet_u, rbuf, state, out, acat, x, t);
    }

    // Final spatio-temporal hypergraph conv over the full sequence (in d_out)
    hyper_es<64><<<B_ * T_, 64, 0, stream>>>(out, lhs, esbuf);
    final_combine<<<(B_ * N_ * 64) / 256, 256, 0, stream>>>(
        out, esbuf, lhs, lht, lgam, lbet);
    // `state` already lives in d_out's tail == `last` output. Done.
}
